// GATEncoder_74302934220997
// MI455X (gfx1250) — matrix operand order repaired and checked by reading, not compiled, not run on hardware
//
#include <hip/hip_runtime.h>

// ---------------------------------------------------------------------------
// GAT encoder for MI455X (gfx1250): 2x GATConv + 3-layer GRU.
// Dense math on bf16 WMMA (v_wmma_f32_16x16x32_bf16), f32 accumulate.
// GEMM A-slabs staged into LDS via TDM tensor_load_to_lds / async-to-LDS.
// ---------------------------------------------------------------------------

typedef __bf16 bf16x16 __attribute__((ext_vector_type(16)));
typedef unsigned short u16x8 __attribute__((ext_vector_type(8)));
typedef float f32x8 __attribute__((ext_vector_type(8)));
typedef unsigned int u32x4 __attribute__((ext_vector_type(4)));
typedef int i32x4 __attribute__((ext_vector_type(4)));
typedef int i32x8 __attribute__((ext_vector_type(8)));

union BFrag { bf16x16 v; u16x8 u[2]; };

#define AS1 __attribute__((address_space(1)))
#define AS3 __attribute__((address_space(3)))

#if __has_builtin(__builtin_amdgcn_tensor_load_to_lds)
#define HAVE_TDM 1
#endif
#if __has_builtin(__builtin_amdgcn_global_load_async_to_lds_b128)
#define HAVE_ASYNC 1
#endif

// problem constants
constexpr int NN  = 32768;    // nodes
constexpr int E0  = 262144;   // raw edges
constexpr int ET  = E0 + NN;  // + self loops
constexpr int IND = 128;      // input dim
constexpr int H1  = 3;        // heads layer 1
constexpr int D1  = 256;      // hidden per head layer 1
constexpr int C1  = H1 * D1;  // 768
constexpr int D2  = 384;      // layer-2 / GRU dim
constexpr int G3  = 3 * D2;   // 1152 stacked gates
constexpr int TT  = 2048;     // seq len
constexpr int BG  = 16;       // graphs (batch)
constexpr int KMAX = 768;     // max GEMM K

__device__ __forceinline__ unsigned short f32_to_bf16_bits(float f) {
    unsigned int b = __float_as_uint(f);
    b += 0x7FFFu + ((b >> 16) & 1u);        // round-to-nearest-even
    return (unsigned short)(b >> 16);
}

// monotonic float<->uint map for atomicMax on floats
__device__ __forceinline__ unsigned int ordered_f32(float f) {
    unsigned int b = __float_as_uint(f);
    return (b & 0x80000000u) ? ~b : (b | 0x80000000u);
}
__device__ __forceinline__ float unordered_f32(unsigned int u) {
    unsigned int b = (u & 0x80000000u) ? (u & 0x7FFFFFFFu) : ~u;
    return __uint_as_float(b);
}

// ---------------------------------------------------------------------------
__global__ void k_zero(unsigned int* __restrict__ p, int n) {
    int i = blockIdx.x * blockDim.x + threadIdx.x;
    if (i < n) p[i] = 0u;
}

__global__ void k_cvt_bf16(const float* __restrict__ s, unsigned short* __restrict__ d, int n) {
    int i = blockIdx.x * blockDim.x + threadIdx.x;
    if (i < n) d[i] = f32_to_bf16_bits(s[i]);
}

// W [K][N] f32 row-major  ->  Wt [N][K] bf16 row-major
__global__ void k_tcvt_bf16(const float* __restrict__ W, unsigned short* __restrict__ Wt,
                            int K, int N) {
    int i = blockIdx.x * blockDim.x + threadIdx.x;
    if (i < N * K) {
        int n = i / K, k = i - n * K;
        Wt[i] = f32_to_bf16_bits(W[(size_t)k * N + n]);
    }
}

// ---------------------------------------------------------------------------
// C[M][N] = A[M][K] * Bt[N][K]^T  (bf16 in, f32 out), optional bias / relu.
// grid.x = M/16; block = (32,8) = 8 waves. The block's 16xK bf16 A-slab is
// staged ONCE into LDS (TDM tensor DMA if available, else async-to-LDS, else
// plain copy); every wave then computes N/16/8 output tiles from the shared
// slab. Fragment layouts per CDNA5 ISA 7.12.2.
__global__ void k_gemm_bf16(const unsigned short* __restrict__ A,
                            const unsigned short* __restrict__ Bt,
                            float* __restrict__ C,
                            int M, int N, int K,
                            const float* __restrict__ bias, int relu) {
    __shared__ unsigned short aA[16 * KMAX];       // 24 KB max
    const int tm  = blockIdx.x;
    const int tid = threadIdx.y * 32 + threadIdx.x;
    const unsigned short* agbase = A + (size_t)tm * 16 * K;   // contiguous 16xK slab

#if defined(HAVE_TDM)
    if (threadIdx.y == 0) {                        // one wave issues the tensor DMA
        const unsigned lds_base = (unsigned)(unsigned long long)(AS3 unsigned short*)aA;
        const unsigned long long ga = (unsigned long long)agbase;
        const unsigned W4 = (unsigned)(8 * K);     // slab length in 4-byte units
        u32x4 g0;
        g0[0] = 1u;                                           // count = 1 descriptor
        g0[1] = lds_base;                                     // lds_addr
        g0[2] = (unsigned)(ga & 0xFFFFFFFFull);               // global_addr[31:0]
        g0[3] = (unsigned)((ga >> 32) & 0x1FFFFFFull) | (2u << 30);  // addr[56:32]|type=2
        i32x8 g1;
        g1[0] = (2 << 16);                                    // data_size = 4 bytes
        g1[1] = (int)((W4 & 0xFFFFu) << 16);                  // tensor_dim0[15:0]
        g1[2] = (int)((W4 >> 16) & 0xFFFFu);                  // tensor_dim0[31:16], td1=0
        g1[3] = (int)((W4 & 0xFFFFu) << 16);                  // tile_dim0 = W4 (1-D tile)
        g1[4] = 0;                                            // tile_dim1/2 unused
        g1[5] = (int)W4;                                      // stride0 (benign for 1-D)
        g1[6] = 0; g1[7] = 0;
        i32x4 gz4 = {0, 0, 0, 0};
        i32x8 gz8 = {0, 0, 0, 0, 0, 0, 0, 0};
        __builtin_amdgcn_tensor_load_to_lds(g0, g1, gz4, gz4, gz8, 0);
        __builtin_amdgcn_s_wait_tensorcnt(0);
    }
#elif defined(HAVE_ASYNC)
    for (int i = tid; i < 2 * K; i += 256) {       // 16-byte chunks of the slab
        const int off = i * 8;
        __builtin_amdgcn_global_load_async_to_lds_b128(
            (AS1 void*)(void*)(agbase + off), (AS3 void*)(aA + off), 0, 0);
    }
#if __has_builtin(__builtin_amdgcn_s_wait_asynccnt)
    __builtin_amdgcn_s_wait_asynccnt(0);
#else
    asm volatile("s_wait_asynccnt 0" ::: "memory");
#endif
#else
    for (int i = tid; i < 2 * K; i += 256) {
        const int off = i * 8;
        *(u16x8*)(aA + off) = *(const u16x8*)(agbase + off);
    }
#endif
    __syncthreads();

    const int lane = threadIdx.x & 31;
    const int l15 = lane & 15, half = lane >> 4;
    const unsigned short* arow = aA + (size_t)l15 * K;

    for (int tn = threadIdx.y; tn < (N >> 4); tn += 8) {
        const unsigned short* brow = Bt + (size_t)(tn * 16 + l15) * K;
        f32x8 acc = {0.f, 0.f, 0.f, 0.f, 0.f, 0.f, 0.f, 0.f};
        for (int kc = 0; kc < K; kc += 32) {
            __builtin_prefetch(brow + kc + 32, 0, 1);   // global_prefetch_b8
            BFrag af, bf;
            // A lane(l15,half): K = 8*half+j (j<8) and 16+8*half+(j-8)
            af.u[0] = *(const u16x8*)(arow + kc + 8 * half);
            af.u[1] = *(const u16x8*)(arow + kc + 16 + 8 * half);
            // B lane = column; K = 8*half+j (j<8) and 16+8*half+(j-8), same as A
            bf.u[0] = *(const u16x8*)(brow + kc + 8 * half);
            bf.u[1] = *(const u16x8*)(brow + kc + 16 + 8 * half);
            acc = __builtin_amdgcn_wmma_f32_16x16x32_bf16(false, af.v, false, bf.v,
                                                          (short)0, acc, false, false);
        }
        const int col = tn * 16 + l15;
        const float bv = bias ? bias[col] : 0.f;
#pragma unroll
        for (int i = 0; i < 8; ++i) {
            float v = acc[i] + bv;
            if (relu) v = fmaxf(v, 0.f);
            C[(size_t)(tm * 16 + i + 8 * half) * N + col] = v;
        }
    }
}

// ---------------------------------------------------------------------------
// alpha_{s,d}[n][h] = dot(hfeat[n, h*D : (h+1)*D], att_{s,d}[h])
__global__ void k_alpha(const float* __restrict__ hfeat,
                        const float* __restrict__ att_s, const float* __restrict__ att_d,
                        float* __restrict__ as, float* __restrict__ ad,
                        int H, int D) {
    int idx = blockIdx.x * blockDim.x + threadIdx.x;
    if (idx >= NN * H) return;
    int n = idx / H, h = idx - n * H;
    const float* hp = hfeat + (size_t)n * H * D + (size_t)h * D;
    const float* sp = att_s + h * D;
    const float* dp = att_d + h * D;
    float s = 0.f, d = 0.f;
    for (int i = 0; i < D; ++i) { float v = hp[i]; s += v * sp[i]; d += v * dp[i]; }
    as[idx] = s; ad[idx] = d;
}

__device__ __forceinline__ void edge_ends(const int* __restrict__ ei, int e, int& src, int& dst) {
    if (e < E0) { src = ei[e]; dst = ei[E0 + e]; }
    else        { src = dst = e - E0; }            // self loop
}

__global__ void k_edge_max(const int* __restrict__ ei,
                           const float* __restrict__ as, const float* __restrict__ ad,
                           unsigned int* __restrict__ emax, int H) {
    int idx = blockIdx.x * blockDim.x + threadIdx.x;
    if (idx >= ET * H) return;
    int e = idx / H, h = idx - e * H;
    int src, dst; edge_ends(ei, e, src, dst);
    float v = as[src * H + h] + ad[dst * H + h];
    v = (v >= 0.f) ? v : 0.2f * v;                 // leaky relu
    atomicMax(&emax[dst * H + h], ordered_f32(v));
}

__global__ void k_edge_exp(const int* __restrict__ ei,
                           const float* __restrict__ as, const float* __restrict__ ad,
                           const unsigned int* __restrict__ emax,
                           float* __restrict__ aE, float* __restrict__ denom, int H) {
    int idx = blockIdx.x * blockDim.x + threadIdx.x;
    if (idx >= ET * H) return;
    int e = idx / H, h = idx - e * H;
    int src, dst; edge_ends(ei, e, src, dst);
    float v = as[src * H + h] + ad[dst * H + h];
    v = (v >= 0.f) ? v : 0.2f * v;
    float ex = expf(v - unordered_f32(emax[dst * H + h]));
    aE[idx] = ex;
    atomicAdd(&denom[dst * H + h], ex);
}

// one wave per (edge, head); lanes stride the feature dim
__global__ void k_scatter(const int* __restrict__ ei,
                          const float* __restrict__ hfeat,
                          const float* __restrict__ aE, const float* __restrict__ denom,
                          float* __restrict__ acc, int H, int D) {
    int gid  = blockIdx.x * blockDim.x + threadIdx.x;
    int wave = gid >> 5, lane = gid & 31;
    if (wave >= ET * H) return;
    int e = wave / H, h = wave - e * H;
    int src, dst; edge_ends(ei, e, src, dst);
    float a = aE[wave] / (denom[dst * H + h] + 1e-16f);
    const float* hp = hfeat + (size_t)src * H * D + (size_t)h * D;
    float*       op = acc   + (size_t)dst * H * D + (size_t)h * D;
    for (int i = lane; i < D; i += 32) atomicAdd(&op[i], hp[i] * a);
}

// out_bf = bf16(relu(acc + bias[col]))
__global__ void k_finalize(const float* __restrict__ acc, const float* __restrict__ bias,
                           unsigned short* __restrict__ out_bf, int Ctot) {
    int idx = blockIdx.x * blockDim.x + threadIdx.x;
    if (idx >= NN * Ctot) return;
    int c = idx % Ctot;
    float v = fmaxf(acc[idx] + bias[c], 0.f);
    out_bf[idx] = f32_to_bf16_bits(v);
}

// ---------------------------------------------------------------------------
// One GRU layer: persistent single workgroup (32 waves, wave32), loops T steps.
// gi   : [NN][G3] f32, node-major (node n = b*T + t) -- precomputed x@w_ih^T
// whh  : [G3][D2] bf16 (L2-resident, re-read each step)
// ys_bf: node-major bf16 output (nullable), ys_f32: f32 output (nullable)
__global__ void __launch_bounds__(1024)
k_gru_layer(const float* __restrict__ gi, const unsigned short* __restrict__ whh,
            const float* __restrict__ b_ih, const float* __restrict__ b_hh,
            unsigned short* __restrict__ ys_bf, float* __restrict__ ys_f32) {
    __shared__ unsigned short h_bf[BG][D2];   // 12 KB : A operand for WMMA
    __shared__ float          gh[BG][G3];     // 72 KB : h @ whh^T per step

    const int tid  = threadIdx.x;
    const int lane = tid & 31, wv = tid >> 5;          // 32 waves
    const int l15 = lane & 15, half = lane >> 4;

    // per-thread registers hold h_prev for its 6 fixed elements (BG*D2 = 6*1024)
    float hreg[6];
#pragma unroll
    for (int k = 0; k < 6; ++k) hreg[k] = 0.f;
    for (int i = tid; i < BG * D2; i += 1024) ((unsigned short*)h_bf)[i] = 0;
    __syncthreads();

    for (int t = 0; t < TT; ++t) {
        // ---- phase A: gh = h @ whh^T   (M=16, N=1152, K=384) ----
        for (int tile = wv; tile < G3 / 16; tile += 32) {
            const unsigned short* brow = whh + (size_t)(tile * 16 + l15) * D2;
            const unsigned short* arow = &h_bf[l15][0];
            f32x8 acc = {0.f, 0.f, 0.f, 0.f, 0.f, 0.f, 0.f, 0.f};
            for (int kc = 0; kc < D2; kc += 32) {
                BFrag af, bf;
                af.u[0] = *(const u16x8*)(arow + kc + 8 * half);
                af.u[1] = *(const u16x8*)(arow + kc + 16 + 8 * half);
                bf.u[0] = *(const u16x8*)(brow + kc + 8 * half);
                bf.u[1] = *(const u16x8*)(brow + kc + 16 + 8 * half);
                acc = __builtin_amdgcn_wmma_f32_16x16x32_bf16(false, af.v, false, bf.v,
                                                              (short)0, acc, false, false);
            }
            const int col = tile * 16 + l15;
#pragma unroll
            for (int i = 0; i < 8; ++i) gh[i + 8 * half][col] = acc[i];
        }
        __syncthreads();

        // ---- phase B: fused gates ----
#pragma unroll
        for (int k = 0; k < 6; ++k) {
            const int i = tid + k * 1024;          // i < BG*D2
            const int b = i / D2, d = i - b * D2;
            const size_t nrow = (size_t)b * TT + t;
            const float* gir = gi + nrow * G3;
            __builtin_prefetch(gir + G3 + d, 0, 1);        // next timestep's gi row
            float i_r = gir[d]          + b_ih[d];
            float i_z = gir[d + D2]     + b_ih[d + D2];
            float i_n = gir[d + 2 * D2] + b_ih[d + 2 * D2];
            float h_r = gh[b][d]          + b_hh[d];
            float h_z = gh[b][d + D2]     + b_hh[d + D2];
            float h_n = gh[b][d + 2 * D2] + b_hh[d + 2 * D2];
            float r  = 1.f / (1.f + expf(-(i_r + h_r)));
            float z  = 1.f / (1.f + expf(-(i_z + h_z)));
            float nn = tanhf(i_n + r * h_n);
            float hnew = (1.f - z) * nn + z * hreg[k];
            hreg[k] = hnew;
            h_bf[b][d] = f32_to_bf16_bits(hnew);
            if (ys_bf)  ys_bf[nrow * D2 + d]  = f32_to_bf16_bits(hnew);
            if (ys_f32) ys_f32[nrow * D2 + d] = hnew;
        }
        __syncthreads();
    }
}

// ---------------------------------------------------------------------------
extern "C" void kernel_launch(void* const* d_in, const int* in_sizes, int n_in,
                              void* d_out, int out_size, void* d_ws, size_t ws_size,
                              hipStream_t stream) {
    (void)in_sizes; (void)n_in; (void)out_size; (void)ws_size;
    const float* x        = (const float*)d_in[0];
    const int*   ei       = (const int*)d_in[1];
    const float* W1       = (const float*)d_in[3];
    const float* att_s1   = (const float*)d_in[4];
    const float* att_d1   = (const float*)d_in[5];
    const float* b1       = (const float*)d_in[6];
    const float* W2       = (const float*)d_in[7];
    const float* att_s2   = (const float*)d_in[8];
    const float* att_d2   = (const float*)d_in[9];
    const float* b2       = (const float*)d_in[10];
    const float* wih      = (const float*)d_in[11];
    const float* whh      = (const float*)d_in[12];
    const float* bih      = (const float*)d_in[13];
    const float* bhh      = (const float*)d_in[14];
    float* out = (float*)d_out;

    // ---- workspace carve-out (256B aligned) ----
    char* w = (char*)d_ws; size_t off = 0;
    auto alloc = [&](size_t bytes) -> void* {
        void* p = w + off; off += (bytes + 255) & ~(size_t)255; return p;
    };
    unsigned short* x_bf    = (unsigned short*)alloc((size_t)NN * IND * 2);
    unsigned short* W1t     = (unsigned short*)alloc((size_t)C1 * IND * 2);
    unsigned short* W2t     = (unsigned short*)alloc((size_t)D2 * C1 * 2);
    unsigned short* wih_bf  = (unsigned short*)alloc((size_t)3 * G3 * D2 * 2);
    unsigned short* whh_bf  = (unsigned short*)alloc((size_t)3 * G3 * D2 * 2);
    float*          h1raw   = (float*)alloc((size_t)NN * C1 * 4);
    float*          as1     = (float*)alloc((size_t)NN * H1 * 4);
    float*          ad1     = (float*)alloc((size_t)NN * H1 * 4);
    unsigned int*   emax1   = (unsigned int*)alloc((size_t)NN * H1 * 4);
    float*          den1    = (float*)alloc((size_t)NN * H1 * 4);
    float*          aE1     = (float*)alloc((size_t)ET * H1 * 4);
    float*          acc1    = (float*)alloc((size_t)NN * C1 * 4);
    unsigned short* h1r_bf  = (unsigned short*)alloc((size_t)NN * C1 * 2);
    float*          h2raw   = (float*)alloc((size_t)NN * D2 * 4);
    float*          as2     = (float*)alloc((size_t)NN * 4);
    float*          ad2     = (float*)alloc((size_t)NN * 4);
    unsigned int*   emax2   = (unsigned int*)alloc((size_t)NN * 4);
    float*          den2    = (float*)alloc((size_t)NN * 4);
    float*          aE2     = (float*)alloc((size_t)ET * 4);
    float*          acc2    = (float*)alloc((size_t)NN * D2 * 4);
    unsigned short* h2r_bf  = (unsigned short*)alloc((size_t)NN * D2 * 2);
    float*          gi      = (float*)alloc((size_t)NN * G3 * 4);
    unsigned short* ys_bf   = (unsigned short*)alloc((size_t)NN * D2 * 2);

    auto cdiv = [](int a, int b) { return (a + b - 1) / b; };
    const dim3 gblk(32, 8, 1);   // 8 waves per block for GEMM
    const int  mblocks = NN / 16;

    // ---- weight / input conversion ----
    k_cvt_bf16<<<cdiv(NN * IND, 256), 256, 0, stream>>>(x, x_bf, NN * IND);
    k_tcvt_bf16<<<cdiv(C1 * IND, 256), 256, 0, stream>>>(W1, W1t, IND, C1);
    k_tcvt_bf16<<<cdiv(D2 * C1, 256), 256, 0, stream>>>(W2, W2t, C1, D2);
    k_cvt_bf16<<<cdiv(3 * G3 * D2, 256), 256, 0, stream>>>(wih, wih_bf, 3 * G3 * D2);
    k_cvt_bf16<<<cdiv(3 * G3 * D2, 256), 256, 0, stream>>>(whh, whh_bf, 3 * G3 * D2);

    // ---- GAT layer 1 ----
    k_gemm_bf16<<<mblocks, gblk, 0, stream>>>(x_bf, W1t, h1raw, NN, C1, IND, nullptr, 0);
    k_alpha<<<cdiv(NN * H1, 256), 256, 0, stream>>>(h1raw, att_s1, att_d1, as1, ad1, H1, D1);
    k_zero<<<cdiv(NN * H1, 256), 256, 0, stream>>>(emax1, NN * H1);
    k_zero<<<cdiv(NN * H1, 256), 256, 0, stream>>>((unsigned int*)den1, NN * H1);
    k_zero<<<cdiv(NN * C1, 256), 256, 0, stream>>>((unsigned int*)acc1, NN * C1);
    k_edge_max<<<cdiv(ET * H1, 256), 256, 0, stream>>>(ei, as1, ad1, emax1, H1);
    k_edge_exp<<<cdiv(ET * H1, 256), 256, 0, stream>>>(ei, as1, ad1, emax1, aE1, den1, H1);
    k_scatter<<<cdiv(ET * H1 * 32, 256), 256, 0, stream>>>(ei, h1raw, aE1, den1, acc1, H1, D1);
    k_finalize<<<cdiv(NN * C1, 256), 256, 0, stream>>>(acc1, b1, h1r_bf, C1);

    // ---- GAT layer 2 ----
    k_gemm_bf16<<<mblocks, gblk, 0, stream>>>(h1r_bf, W2t, h2raw, NN, D2, C1, nullptr, 0);
    k_alpha<<<cdiv(NN, 256), 256, 0, stream>>>(h2raw, att_s2, att_d2, as2, ad2, 1, D2);
    k_zero<<<cdiv(NN, 256), 256, 0, stream>>>(emax2, NN);
    k_zero<<<cdiv(NN, 256), 256, 0, stream>>>((unsigned int*)den2, NN);
    k_zero<<<cdiv(NN * D2, 256), 256, 0, stream>>>((unsigned int*)acc2, NN * D2);
    k_edge_max<<<cdiv(ET, 256), 256, 0, stream>>>(ei, as2, ad2, emax2, 1);
    k_edge_exp<<<cdiv(ET, 256), 256, 0, stream>>>(ei, as2, ad2, emax2, aE2, den2, 1);
    k_scatter<<<cdiv(ET * 32, 256), 256, 0, stream>>>(ei, h2raw, aE2, den2, acc2, 1, D2);
    k_finalize<<<cdiv(NN * D2, 256), 256, 0, stream>>>(acc2, b2, h2r_bf, D2);

    // ---- GRU: 3 layers ----
    for (int l = 0; l < 3; ++l) {
        const unsigned short* in_bf = (l == 0) ? h2r_bf : ys_bf;
        k_gemm_bf16<<<mblocks, gblk, 0, stream>>>(
            in_bf, wih_bf + (size_t)l * G3 * D2, gi, NN, G3, D2, nullptr, 0);
        k_gru_layer<<<1, 1024, 0, stream>>>(
            gi, whh_bf + (size_t)l * G3 * D2, bih + (size_t)l * G3, bhh + (size_t)l * G3,
            (l < 2) ? ys_bf : nullptr, (l == 2) ? out : nullptr);
    }
}
